// MultiHeadAttention_43542378447001
// MI455X (gfx1250) — compile-verified, matrix operand order checked by reading, not run on hardware
//
#include <hip/hip_runtime.h>
#include <stdint.h>

// ---------------- types ----------------
typedef __bf16 bf16_t;
typedef __attribute__((ext_vector_type(16))) __bf16 v16bf;
typedef __attribute__((ext_vector_type(8)))  __bf16 v8bf;
typedef __attribute__((ext_vector_type(8)))  float  v8f;
typedef __attribute__((ext_vector_type(4)))  uint32_t v4u;
typedef __attribute__((ext_vector_type(8)))  uint32_t v8u;

union U16 { v16bf v; v8bf h[2]; };

#define EMBED   2048
#define NHEADS  16
#define HDIM    128
#define SEQ     2048
#define BATCH   2
#define MROWS   (BATCH*SEQ)          // 4096
#define QKV_N   (3*EMBED)            // 6144

// element counts for workspace carving
#define XB_E    ((size_t)MROWS*EMBED)           // 8388608
#define WQKV_E  ((size_t)QKV_N*EMBED)           // 12582912
#define WOUT_E  ((size_t)EMBED*EMBED)           // 4194304
#define QKV1_E  ((size_t)BATCH*NHEADS*SEQ*HDIM) // 8388608

// ---------------- convert kernels ----------------
__global__ void cvt_copy_bf16(const float* __restrict__ src, bf16_t* __restrict__ dst, int n) {
    int i = blockIdx.x * blockDim.x + threadIdx.x;
    if (i < n) dst[i] = (bf16_t)src[i];
}

// src [rows x cols] row-major fp32  ->  dst [cols x rows] row-major bf16
__global__ void cvt_transpose_bf16(const float* __restrict__ src, bf16_t* __restrict__ dst,
                                   int rows, int cols) {
    int i = blockIdx.x * blockDim.x + threadIdx.x;
    int n = rows * cols;
    if (i < n) {
        int c = i / rows;
        int k = i - c * rows;
        dst[i] = (bf16_t)src[(size_t)k * cols + c];
    }
}

// ---------------- TDM helper: 2D tile load into LDS ----------------
// D# group0 (count=1, type=2) + group1 (data_size=2B, dims in elements);
// tensor_load_to_lds 2-group form (tensors up to 2D).
__device__ __forceinline__ void tdm_load_2d(uint32_t lds_off, const void* gaddr,
                                            uint32_t tdim0, uint32_t tdim1,
                                            uint32_t tile0, uint32_t tile1,
                                            uint32_t stride0) {
    uint64_t ga = (uint64_t)(uintptr_t)gaddr;
    v4u g0;
    g0[0] = 1u;                                              // count=1, user mode
    g0[1] = lds_off;                                         // lds_addr (bytes)
    g0[2] = (uint32_t)ga;                                    // global_addr[31:0]
    g0[3] = ((uint32_t)(ga >> 32) & 0x01FFFFFFu)             // global_addr[56:32]
            | 0x80000000u;                                   // type=2 ("image")
    v8u g1;
    g1[0] = 0x00010000u;                                     // data_size=1 -> 2 bytes
    g1[1] = (tdim0 & 0xFFFFu) << 16;                         // tensor_dim0 lo16 @bit48
    g1[2] = (tdim0 >> 16) | ((tdim1 & 0xFFFFu) << 16);       // dim0 hi / dim1 lo
    g1[3] = (tdim1 >> 16) | (tile0 << 16);                   // dim1 hi / tile_dim0
    g1[4] = tile1;                                           // tile_dim1 (tile_dim2=0)
    g1[5] = stride0;                                         // tensor_dim0_stride lo32
    g1[6] = 0u;
    g1[7] = 0u;
    asm volatile("tensor_load_to_lds %0, %1" :: "s"(g0), "s"(g1) : "memory");
}

// ---------------- QKV GEMM: async-LDS staged, double buffered ----------------
// Block tile 128M x 64N, 8 waves; each wave computes 16M x 64N.
// scatter: n<2048 -> Q[b,h,s,d] ; n<4096 -> K[b,h,s,d] ; else Vt[b,h,d,s]
__global__ void __launch_bounds__(256)
qkv_gemm_kernel(const bf16_t* __restrict__ xb, const bf16_t* __restrict__ wT,
                bf16_t* __restrict__ Qb, bf16_t* __restrict__ Kb,
                bf16_t* __restrict__ Vtb) {
    __shared__ __align__(16) bf16_t ldsA[2][128 * 32];   // 2 x 8 KB
    __shared__ __align__(16) bf16_t ldsB[2][64 * 32];    // 2 x 4 KB

    const int tid  = threadIdx.x;
    const int lane = tid & 31, wave = tid >> 5;
    const int hf   = lane >> 4, r = lane & 15;
    const int NB   = QKV_N / 64;                         // 96
    const int mb   = blockIdx.x / NB, nb = blockIdx.x % NB;
    const int m0   = mb * 128, n0 = nb * 64;

    const uint32_t aBase = (uint32_t)(uintptr_t)(&ldsA[0][0]);
    const uint32_t bBase = (uint32_t)(uintptr_t)(&ldsB[0][0]);

    // stage the 32-wide K slice k0 into buffer buf (async, ASYNCcnt-tracked)
    auto stage = [&](int k0, int buf) {
        const uint32_t aOff = aBase + (uint32_t)buf * (128 * 32 * 2);
        const uint32_t bOff = bBase + (uint32_t)buf * (64 * 32 * 2);
        int c0 = tid, c1 = tid + 256;                    // A: 512 x 16B chunks
        uint64_t ga0 = (uint64_t)(uintptr_t)(xb + (size_t)(m0 + (c0 >> 2)) * EMBED + k0 + (c0 & 3) * 8);
        uint64_t ga1 = (uint64_t)(uintptr_t)(xb + (size_t)(m0 + (c1 >> 2)) * EMBED + k0 + (c1 & 3) * 8);
        asm volatile("global_load_async_to_lds_b128 %0, %1, off"
                     :: "v"(aOff + (uint32_t)c0 * 16), "v"(ga0) : "memory");
        asm volatile("global_load_async_to_lds_b128 %0, %1, off"
                     :: "v"(aOff + (uint32_t)c1 * 16), "v"(ga1) : "memory");
        if (tid < 128) {                                 // B: 128 x 16B chunks
            uint64_t gb = (uint64_t)(uintptr_t)(wT + (size_t)(n0 + (tid >> 2)) * EMBED + k0 + (tid & 3) * 8);
            asm volatile("global_load_async_to_lds_b128 %0, %1, off"
                         :: "v"(bOff + (uint32_t)tid * 16), "v"(gb) : "memory");
        }
    };

    v8f acc[4] = {};
    stage(0, 0);
    for (int k0 = 0; k0 < EMBED; k0 += 32) {
        const int  buf  = (k0 >> 5) & 1;
        const bool more = (k0 + 32) < EMBED;
        if (more) {
            stage(k0 + 32, buf ^ 1);
            // wait for the PREVIOUS slice only (the ops just issued stay in flight)
            if (tid < 128) asm volatile("s_wait_asynccnt 0x3" ::: "memory");
            else           asm volatile("s_wait_asynccnt 0x2" ::: "memory");
        } else {
            asm volatile("s_wait_asynccnt 0x0" ::: "memory");
        }
        __syncthreads();

        const bf16_t* As = &ldsA[buf][0];
        const bf16_t* Bs = &ldsB[buf][0];
        // hoist ALL fragments, then issue WMMAs back-to-back (XDL-friendly)
        U16 a;
        a.h[0] = *(const v8bf*)(As + (wave * 16 + r) * 32 + hf * 8);
        a.h[1] = *(const v8bf*)(As + (wave * 16 + r) * 32 + 16 + hf * 8);
        U16 bm[4];
#pragma unroll
        for (int j = 0; j < 4; ++j) {
            bm[j].h[0] = *(const v8bf*)(Bs + (j * 16 + r) * 32 + hf * 8);
            bm[j].h[1] = *(const v8bf*)(Bs + (j * 16 + r) * 32 + 16 + hf * 8);
        }
#pragma unroll
        for (int j = 0; j < 4; ++j)
            acc[j] = __builtin_amdgcn_wmma_f32_16x16x32_bf16(
                false, a.v, false, bm[j].v, (short)0, acc[j], false, false);
        __syncthreads();
    }

#pragma unroll
    for (int j = 0; j < 4; ++j) {
#pragma unroll
        for (int i = 0; i < 8; ++i) {
            int row = m0 + wave * 16 + hf * 8 + i;
            int col = n0 + j * 16 + r;
            float v = acc[j][i];
            int b = row >> 11, s = row & 2047;
            int which = col >> 11;
            int hcol  = (col & 2047) >> 7;
            int d     = col & 127;
            int bh    = b * NHEADS + hcol;
            if (which == 0)
                Qb[((size_t)bh * SEQ + s) * HDIM + d] = (bf16_t)v;
            else if (which == 1)
                Kb[((size_t)bh * SEQ + s) * HDIM + d] = (bf16_t)v;
            else
                Vtb[((size_t)bh * HDIM + d) * SEQ + s] = (bf16_t)v;
        }
    }
}

// ---------------- RoPE (+ fold 1/sqrt(D) into Q) ----------------
__global__ void rope_kernel(bf16_t* __restrict__ Qb, bf16_t* __restrict__ Kb) {
    int idx = blockIdx.x * blockDim.x + threadIdx.x;     // over B*H*S*64
    if (idx >= BATCH * NHEADS * SEQ * 64) return;
    int d2 = idx & 63;
    int s  = (idx >> 6) & (SEQ - 1);
    int bh = idx >> 17;
    size_t base = ((size_t)bh * SEQ + s) * HDIM;

    float inv = __expf(-9.210340371976184f * (float)d2 * (1.0f / 64.0f)); // 10000^{-d2/64}
    float fr  = (float)s * inv;
    float c = __cosf(fr), sn = __sinf(fr);
    const float sc = 0.08838834764831845f;               // 1/sqrt(128)

    float q1 = (float)Qb[base + d2], q2 = (float)Qb[base + d2 + 64];
    Qb[base + d2]      = (bf16_t)((q1 * c - q2 * sn) * sc);
    Qb[base + d2 + 64] = (bf16_t)((q2 * c + q1 * sn) * sc);

    float k1 = (float)Kb[base + d2], k2 = (float)Kb[base + d2 + 64];
    Kb[base + d2]      = (bf16_t)(k1 * c - k2 * sn);
    Kb[base + d2 + 64] = (bf16_t)(k2 * c + k1 * sn);
}

// ---------------- half-wave (16-lane) reductions ----------------
__device__ __forceinline__ float hred_max(float v) {
#pragma unroll
    for (int m = 1; m < 16; m <<= 1) v = fmaxf(v, __shfl_xor(v, m, 32));
    return v;
}
__device__ __forceinline__ float hred_sum(float v) {
#pragma unroll
    for (int m = 1; m < 16; m <<= 1) v += __shfl_xor(v, m, 32);
    return v;
}

// ---------------- causal flash attention ----------------
// grid (16 q-blocks, 32 bh); 8 waves/block; wave = 16 queries, full D=128
__global__ void attn_kernel(const bf16_t* __restrict__ Qb, const bf16_t* __restrict__ Kb,
                            const bf16_t* __restrict__ Vtb, bf16_t* __restrict__ ctx) {
    __shared__ __align__(16) bf16_t lds_p[8][16][32];

    const int lane = threadIdx.x & 31;
    const int wave = threadIdx.x >> 5;
    const int hf   = lane >> 4, r = lane & 15;
    const int bh   = blockIdx.y;
    const int b    = bh >> 4, h = bh & 15;
    const int q0   = blockIdx.x * 128 + wave * 16;

    const bf16_t* Qh = Qb  + (size_t)bh * SEQ * HDIM;
    const bf16_t* Kh = Kb  + (size_t)bh * SEQ * HDIM;
    const bf16_t* Vh = Vtb + (size_t)bh * HDIM * SEQ;

    // A-matrix: 16 queries x 128 d (4 chunks of 32)
    U16 qa[4];
    {
        const bf16_t* qrow = Qh + (size_t)(q0 + r) * HDIM;
#pragma unroll
        for (int c = 0; c < 4; ++c) {
            qa[c].h[0] = *(const v8bf*)(qrow + c * 32 + hf * 8);
            qa[c].h[1] = *(const v8bf*)(qrow + c * 32 + 16 + hf * 8);
        }
    }

    v8f   acco[8] = {};
    float mrow[8], lrow[8];
#pragma unroll
    for (int i = 0; i < 8; ++i) { mrow[i] = -1.0e30f; lrow[i] = 0.0f; }

    const int kend = q0 + 16;                       // keys needed: k <= q0+15
    for (int kb = 0; kb < kend; kb += 32) {
        // prefetch next key block's K/V rows while we compute this one
        if (kb + 32 < kend) {
            __builtin_prefetch(Kh + (size_t)(kb + 32 + r + hf * 16) * HDIM, 0, 1);
            __builtin_prefetch(Vh + (size_t)(lane * 4) * SEQ + kb + 32, 0, 1);
        }

        // ---- scores: load all 8 K fragments, then 8 WMMAs back-to-back ----
        U16 kf[2][4];
#pragma unroll
        for (int t = 0; t < 2; ++t) {
            const bf16_t* krow = Kh + (size_t)(kb + t * 16 + r) * HDIM;
#pragma unroll
            for (int c = 0; c < 4; ++c) {
                kf[t][c].h[0] = *(const v8bf*)(krow + c * 32 + hf * 8);
                kf[t][c].h[1] = *(const v8bf*)(krow + c * 32 + 16 + hf * 8);
            }
        }
        v8f accs[2] = {};
#pragma unroll
        for (int c = 0; c < 4; ++c) {               // interleave t0/t1: dep distance 2
#pragma unroll
            for (int t = 0; t < 2; ++t)
                accs[t] = __builtin_amdgcn_wmma_f32_16x16x32_bf16(
                    false, qa[c].v, false, kf[t][c].v, (short)0, accs[t], false, false);
        }

        // ---- online softmax over this 32-key block ----
#pragma unroll
        for (int i = 0; i < 8; ++i) {
            int row = q0 + hf * 8 + i;
            float s0 = (kb + r      <= row) ? accs[0][i] : -1.0e30f;
            float s1 = (kb + 16 + r <= row) ? accs[1][i] : -1.0e30f;
            float mloc = hred_max(fmaxf(s0, s1));
            float mnew = fmaxf(mrow[i], mloc);
            float p0 = __expf(s0 - mnew);
            float p1 = __expf(s1 - mnew);
            float psum = hred_sum(p0 + p1);
            float alpha = __expf(mrow[i] - mnew);
            lrow[i] = lrow[i] * alpha + psum;
            mrow[i] = mnew;
#pragma unroll
            for (int dt = 0; dt < 8; ++dt) acco[dt][i] *= alpha;
            lds_p[wave][hf * 8 + i][r]      = (bf16_t)p0;
            lds_p[wave][hf * 8 + i][16 + r] = (bf16_t)p1;
        }

        // ---- reload P in A-layout (16q x 32k) via LDS ----
        U16 pa;
        pa.h[0] = *(const v8bf*)(&lds_p[wave][r][hf * 8]);
        pa.h[1] = *(const v8bf*)(&lds_p[wave][r][16 + hf * 8]);

        // ---- P @ V: load all 8 V fragments, then 8 independent WMMAs ----
        U16 vf[8];
#pragma unroll
        for (int dt = 0; dt < 8; ++dt) {
            const bf16_t* vrow = Vh + (size_t)(dt * 16 + r) * SEQ + kb;
            vf[dt].h[0] = *(const v8bf*)(vrow + hf * 8);
            vf[dt].h[1] = *(const v8bf*)(vrow + 16 + hf * 8);
        }
#pragma unroll
        for (int dt = 0; dt < 8; ++dt)
            acco[dt] = __builtin_amdgcn_wmma_f32_16x16x32_bf16(
                false, pa.v, false, vf[dt].v, (short)0, acco[dt], false, false);
    }

    // ---- epilogue: normalize, write ctx[b, s, h*128+d] ----
#pragma unroll
    for (int dt = 0; dt < 8; ++dt) {
#pragma unroll
        for (int i = 0; i < 8; ++i) {
            int row = q0 + hf * 8 + i;
            int col = h * HDIM + dt * 16 + r;
            float o = acco[dt][i] / lrow[i];
            ctx[((size_t)b * SEQ + row) * EMBED + col] = (bf16_t)o;
        }
    }
}

// ---------------- output projection: TDM-staged GEMM ----------------
// out[4096 x 2048] = ctx @ w_out ; block tile 128M x 64N; panels staged by TDM
__global__ void __launch_bounds__(256)
out_gemm_kernel(const bf16_t* __restrict__ ctx, const bf16_t* __restrict__ wT,
                float* __restrict__ out) {
    __shared__ __align__(16) bf16_t ldsA[128 * 32];      // 8 KB
    __shared__ __align__(16) bf16_t ldsB[64 * 32];       // 4 KB

    const int tid  = threadIdx.x;
    const int lane = tid & 31, wave = tid >> 5;
    const int hf   = lane >> 4, r = lane & 15;
    const int NB   = EMBED / 64;                         // 32
    const int mb   = blockIdx.x / NB, nb = blockIdx.x % NB;
    const int m0   = mb * 128, n0 = nb * 64;

    const uint32_t aOff = (uint32_t)(uintptr_t)(&ldsA[0]);
    const uint32_t bOff = (uint32_t)(uintptr_t)(&ldsB[0]);

    v8f acc[4] = {};
    for (int k0 = 0; k0 < EMBED; k0 += 32) {
        __syncthreads();                                 // previous compute done
        if (wave == 0) {
            // A panel: 128 rows x 32 k from ctx[4096 x 2048]
            tdm_load_2d(aOff, ctx + (size_t)m0 * EMBED + k0,
                        EMBED, MROWS, 32, 128, EMBED);
            // B panel: 64 rows x 32 k from wT[2048 x 2048]
            tdm_load_2d(bOff, wT + (size_t)n0 * EMBED + k0,
                        EMBED, EMBED, 32, 64, EMBED);
            __builtin_amdgcn_s_wait_tensorcnt(0);
        }
        __syncthreads();                                 // panels visible to all

        U16 a;
        a.h[0] = *(const v8bf*)(&ldsA[(wave * 16 + r) * 32 + hf * 8]);
        a.h[1] = *(const v8bf*)(&ldsA[(wave * 16 + r) * 32 + 16 + hf * 8]);
        U16 bm[4];
#pragma unroll
        for (int j = 0; j < 4; ++j) {
            bm[j].h[0] = *(const v8bf*)(&ldsB[(j * 16 + r) * 32 + hf * 8]);
            bm[j].h[1] = *(const v8bf*)(&ldsB[(j * 16 + r) * 32 + 16 + hf * 8]);
        }
#pragma unroll
        for (int j = 0; j < 4; ++j)
            acc[j] = __builtin_amdgcn_wmma_f32_16x16x32_bf16(
                false, a.v, false, bm[j].v, (short)0, acc[j], false, false);
    }

#pragma unroll
    for (int j = 0; j < 4; ++j) {
#pragma unroll
        for (int i = 0; i < 8; ++i) {
            int row = m0 + wave * 16 + hf * 8 + i;
            int col = n0 + j * 16 + r;
            out[(size_t)row * EMBED + col] = acc[j][i];
        }
    }
}

// ---------------- launcher ----------------
extern "C" void kernel_launch(void* const* d_in, const int* in_sizes, int n_in,
                              void* d_out, int out_size, void* d_ws, size_t ws_size,
                              hipStream_t stream) {
    const float* x     = (const float*)d_in[0];
    const float* w_qkv = (const float*)d_in[1];
    const float* w_out = (const float*)d_in[2];
    float* out = (float*)d_out;

    bf16_t* ws    = (bf16_t*)d_ws;
    bf16_t* xb    = ws;
    bf16_t* wqkvT = xb    + XB_E;
    bf16_t* woutT = wqkvT + WQKV_E;
    bf16_t* Qb    = woutT + WOUT_E;
    bf16_t* Kb    = Qb    + QKV1_E;
    bf16_t* Vtb   = Kb    + QKV1_E;
    bf16_t* ctx   = Vtb   + QKV1_E;

    // 1) fp32 -> bf16 conversions (weights pre-transposed for WMMA B-layout)
    cvt_copy_bf16<<<(int)((XB_E + 255) / 256), 256, 0, stream>>>(x, xb, (int)XB_E);
    cvt_transpose_bf16<<<(int)((WQKV_E + 255) / 256), 256, 0, stream>>>(w_qkv, wqkvT, EMBED, QKV_N);
    cvt_transpose_bf16<<<(int)((WOUT_E + 255) / 256), 256, 0, stream>>>(w_out, woutT, EMBED, EMBED);

    // 2) QKV projection (async-LDS staged; scatter Q/K row-major, V transposed)
    qkv_gemm_kernel<<<(MROWS / 128) * (QKV_N / 64), 256, 0, stream>>>(xb, wqkvT, Qb, Kb, Vtb);

    // 3) RoPE on Q (scaled) and K
    rope_kernel<<<(BATCH * NHEADS * SEQ * 64) / 256, 256, 0, stream>>>(Qb, Kb);

    // 4) causal flash attention
    attn_kernel<<<dim3(SEQ / 128, BATCH * NHEADS), 256, 0, stream>>>(Qb, Kb, Vtb, ctx);

    // 5) output projection (TDM-staged) -> fp32
    out_gemm_kernel<<<(MROWS / 128) * (EMBED / 64), 256, 0, stream>>>(ctx, woutT, out);
}
